// MultQueryAttention_88038239634076
// MI455X (gfx1250) — matrix operand order repaired and checked by reading, not compiled, not run on hardware
//
#include <hip/hip_runtime.h>
#include <hip/hip_bf16.h>

typedef _Float16 half_t;
typedef __attribute__((ext_vector_type(8)))  _Float16 v8h;
typedef __attribute__((ext_vector_type(16))) _Float16 v16h;
typedef __attribute__((ext_vector_type(8)))  float    v8f;
typedef int v4i __attribute__((vector_size(16)));

#define HIDDEN 1024
#define HEADS  16
#define HDIM   64
#define BSZ    2
#define SEQ    2048
#define MTOT   (BSZ*SEQ)

// ---------- CDNA5 async global->LDS staging (guarded; falls back to VGPR path) ----------
#if defined(__has_builtin)
#if __has_builtin(__builtin_amdgcn_global_load_async_to_lds_b128)
#define USE_ASYNC_LDS 1
#endif
#endif

typedef __attribute__((address_space(1))) v4i gv4i_t;
typedef __attribute__((address_space(3))) v4i lv4i_t;

// copy 16 bytes per lane: global -> LDS
__device__ __forceinline__ void stage16(const half_t* g, half_t* l) {
#ifdef USE_ASYNC_LDS
  __builtin_amdgcn_global_load_async_to_lds_b128(
      (gv4i_t*)(unsigned long long)g,
      (lv4i_t*)(unsigned int)(unsigned long long)l,   // LDS byte addr = low 32 bits (aperture rule)
      0, 0);
#else
  *(v8h*)l = *(const v8h*)g;
#endif
}

__device__ __forceinline__ void wait_async() {
#ifdef USE_ASYNC_LDS
#if __has_builtin(__builtin_amdgcn_s_wait_asynccnt)
  __builtin_amdgcn_s_wait_asynccnt(0);
#else
  asm volatile("s_wait_asynccnt 0x0" ::: "memory");
#endif
#endif
}

// ---------- helpers ----------
__device__ __forceinline__ v16h mk16(v8h lo, v8h hi) {
  v16h r;
#pragma unroll
  for (int i = 0; i < 8; ++i) { r[i] = lo[i]; r[8 + i] = hi[i]; }
  return r;
}

// ---------- fp32 -> fp16 conversion ----------
__global__ void cvt_f32_f16(const float* __restrict__ in, half_t* __restrict__ out, int n) {
  int i = blockIdx.x * blockDim.x + threadIdx.x;
  int stride = gridDim.x * blockDim.x;
  for (; i < n; i += stride) out[i] = (half_t)in[i];
}

// ---------- tiled WMMA GEMM: C[M,N] = A[M,K] @ B[K,N] + bias ----------
// TK=64, double-buffered LDS: one barrier per K step, loads overlap 8 WMMAs.
#define TM 128
#define TN 64
#define TK 64
#define PAD 8

__global__ __launch_bounds__(256) void gemm_f16_wmma(
    const half_t* __restrict__ A, const half_t* __restrict__ B,
    const float* __restrict__ bias,
    half_t* __restrict__ Ch, float* __restrict__ Cf,
    int M, int N, int K, int out_f32)
{
  __shared__ __align__(16) half_t As[2][TM][TK + PAD];   // [m][k]
  __shared__ __align__(16) half_t Bs[2][TN][TK + PAD];   // transposed [n][k]

  const int tid  = threadIdx.x;
  const int lane = tid & 31;
  const int wid  = tid >> 5;
  const int wm   = wid & 3;        // 4 wave rows * 32 = 128
  const int wn   = wid >> 2;       // 2 wave cols * 32 = 64
  const int mbase = blockIdx.x * TM;
  const int nbase = blockIdx.y * TN;

  const int lm   = lane & 15;
  const int ka8  = (lane >> 4) * 8;    // A-fragment K sub-offset
  const int kb16 = (lane >> 4) * 16;   // unused; B fragment also uses ka8
  const int hi8  = (lane >> 4) * 8;    // C row offset

  v8f acc[2][2] = {};

  auto load_tiles = [&](int step, int buf) {
    const int kb = step * TK;
    // A tile: 128x64 halfs = 1024 chunks of 8 -> 4 per thread (async path)
#pragma unroll
    for (int i = 0; i < 4; ++i) {
      int chunk = tid + i * 256;
      int r  = chunk >> 3;             // 8 chunks per row
      int cc = (chunk & 7) * 8;
      stage16(A + (size_t)(mbase + r) * K + kb + cc, &As[buf][r][cc]);
    }
    // B tile: 64(K) x 64(N) halfs = 512 chunks -> 2 per thread, transposed store
#pragma unroll
    for (int i = 0; i < 2; ++i) {
      int chunk = tid + i * 256;
      int kk = chunk >> 3;
      int nc = (chunk & 7) * 8;
      v8h bv = *(const v8h*)(B + (size_t)(kb + kk) * N + nbase + nc);
#pragma unroll
      for (int j = 0; j < 8; ++j) Bs[buf][nc + j][kk] = bv[j];
    }
  };

  const int nsteps = K / TK;

  load_tiles(0, 0);
  wait_async();
  __syncthreads();

  for (int step = 0; step < nsteps; ++step) {
    const int buf = step & 1;
    if (step + 1 < nsteps) load_tiles(step + 1, buf ^ 1);   // prefetch next tile

    v16h af[2][2], bf[2][2];
#pragma unroll
    for (int mt = 0; mt < 2; ++mt)
#pragma unroll
      for (int c = 0; c < 2; ++c) {
        const half_t* ap = &As[buf][wm * 32 + mt * 16 + lm][c * 32 + ka8];
        af[mt][c] = mk16(*(const v8h*)ap, *(const v8h*)(ap + 16));
      }
#pragma unroll
    for (int nt = 0; nt < 2; ++nt)
#pragma unroll
      for (int c = 0; c < 2; ++c) {
        const half_t* bp = &Bs[buf][wn * 32 + nt * 16 + lm][c * 32 + ka8];
        bf[nt][c] = mk16(*(const v8h*)bp, *(const v8h*)(bp + 16));
      }
#pragma unroll
    for (int mt = 0; mt < 2; ++mt)
#pragma unroll
      for (int nt = 0; nt < 2; ++nt)
#pragma unroll
        for (int c = 0; c < 2; ++c)
          acc[mt][nt] = __builtin_amdgcn_wmma_f32_16x16x32_f16(
              false, af[mt][c], false, bf[nt][c], (short)0, acc[mt][nt], false, false);

    wait_async();
    __syncthreads();
  }

  // ---- epilogue: bias + store (f16 or f32) ----
#pragma unroll
  for (int mt = 0; mt < 2; ++mt)
#pragma unroll
    for (int nt = 0; nt < 2; ++nt) {
      int col = nbase + wn * 32 + nt * 16 + lm;
      float bsv = bias ? bias[col] : 0.0f;
#pragma unroll
      for (int r = 0; r < 8; ++r) {
        int row = mbase + wm * 32 + mt * 16 + r + hi8;
        float val = acc[mt][nt][r] + bsv;
        if (out_f32) Cf[(size_t)row * N + col] = val;
        else         Ch[(size_t)row * N + col] = (half_t)val;
      }
    }
}

// ---------- flash-style multi-query attention ----------
// grid: (SEQ/64, HEADS, BSZ), block: 128 threads (4 waves, 16 q-rows each)
#define KT 64

__global__ __launch_bounds__(128) void mqa_flash_wmma(
    const half_t* __restrict__ q, const half_t* __restrict__ k,
    const half_t* __restrict__ v, half_t* __restrict__ ctx, int S)
{
  __shared__ __align__(16) half_t Ks[KT][HDIM + 8];      // [kpos][d]  (natural == B-layout for QK^T)
  __shared__ __align__(16) half_t Vs[HDIM][KT + 8];      // [d][kpos]  (transposed for PV)
  __shared__ __align__(16) half_t Ps[4][16][KT + 8];     // per-wave P tile

  const int tid  = threadIdx.x;
  const int lane = tid & 31;
  const int wid  = tid >> 5;
  const int lm   = lane & 15;
  const int ka8  = (lane >> 4) * 8;
  const int kb16 = (lane >> 4) * 16;
  const int hi8  = (lane >> 4) * 8;

  const int h = blockIdx.y;
  const size_t bbase = (size_t)blockIdx.z * S;
  const int qrow0 = blockIdx.x * 64 + wid * 16;

  // Q fragments (head-dim split into 2 K-chunks of 32), softmax scale folded in
  v16h qf[2];
  {
    const half_t* qp = q + ((bbase + qrow0 + lm) * HEADS + h) * (size_t)HDIM;
#pragma unroll
    for (int c = 0; c < 2; ++c) {
      v8h lo = *(const v8h*)(qp + c * 32 + ka8);
      v8h hi = *(const v8h*)(qp + c * 32 + 16 + ka8);
#pragma unroll
      for (int j = 0; j < 8; ++j) {
        lo[j] = lo[j] * (half_t)0.125f;   // 1/sqrt(64)
        hi[j] = hi[j] * (half_t)0.125f;
      }
      qf[c] = mk16(lo, hi);
    }
  }

  v8f oacc[4] = {};
  float mi[8], li[8];
#pragma unroll
  for (int r = 0; r < 8; ++r) { mi[r] = -3.0e38f; li[r] = 0.0f; }

  for (int kt = 0; kt < S; kt += KT) {
    // ---- stage K (async, row-major) and V (transposed) tiles ----
#pragma unroll
    for (int i = 0; i < 4; ++i) {
      int chunk = tid + i * 128;          // 512 chunks of 8 halfs
      int r  = chunk >> 3;                // key position 0..63
      int cc = (chunk & 7) * 8;           // d offset
      stage16(k + (bbase + kt + r) * (size_t)HDIM + cc, &Ks[r][cc]);
      v8h vv = *(const v8h*)(v + (bbase + kt + r) * (size_t)HDIM + cc);
#pragma unroll
      for (int j = 0; j < 8; ++j) Vs[cc + j][r] = vv[j];
    }
    wait_async();
    __syncthreads();

    // ---- S = (Q*scale) @ K^T : 4 n-tiles x 2 K-chunks ----
    v8f sc[4];
#pragma unroll
    for (int nt = 0; nt < 4; ++nt) {
      v8f s = {};
#pragma unroll
      for (int c = 0; c < 2; ++c) {
        const half_t* bp = &Ks[nt * 16 + lm][c * 32 + ka8];
        v16h bf = mk16(*(const v8h*)bp, *(const v8h*)(bp + 16));
        s = __builtin_amdgcn_wmma_f32_16x16x32_f16(false, qf[c], false, bf,
                                                   (short)0, s, false, false);
      }
      sc[nt] = s;
    }

    // ---- online softmax (rows live across 16 lanes of each acc register) ----
#pragma unroll
    for (int r = 0; r < 8; ++r) {
      float tmax = sc[0][r];
#pragma unroll
      for (int nt = 1; nt < 4; ++nt) tmax = fmaxf(tmax, sc[nt][r]);
      tmax = fmaxf(tmax, __shfl_xor(tmax, 1));
      tmax = fmaxf(tmax, __shfl_xor(tmax, 2));
      tmax = fmaxf(tmax, __shfl_xor(tmax, 4));
      tmax = fmaxf(tmax, __shfl_xor(tmax, 8));
      float mnew = fmaxf(mi[r], tmax);
      float corr = __expf(mi[r] - mnew);
      float rsum = 0.0f;
#pragma unroll
      for (int nt = 0; nt < 4; ++nt) {
        float p = __expf(sc[nt][r] - mnew);
        rsum += p;
        Ps[wid][r + hi8][nt * 16 + lm] = (half_t)p;
      }
      rsum += __shfl_xor(rsum, 1);
      rsum += __shfl_xor(rsum, 2);
      rsum += __shfl_xor(rsum, 4);
      rsum += __shfl_xor(rsum, 8);
      li[r] = li[r] * corr + rsum;
      mi[r] = mnew;
#pragma unroll
      for (int dt = 0; dt < 4; ++dt) oacc[dt][r] *= corr;
    }

    // ---- O += P @ V : A-fragments from per-wave LDS, B from transposed V ----
    v16h pf[2];
#pragma unroll
    for (int c = 0; c < 2; ++c) {
      const half_t* pp = &Ps[wid][lm][c * 32 + ka8];
      pf[c] = mk16(*(const v8h*)pp, *(const v8h*)(pp + 16));
    }
#pragma unroll
    for (int dt = 0; dt < 4; ++dt)
#pragma unroll
      for (int c = 0; c < 2; ++c) {
        const half_t* bp = &Vs[dt * 16 + lm][c * 32 + ka8];
        v16h bf = mk16(*(const v8h*)bp, *(const v8h*)(bp + 16));
        oacc[dt] = __builtin_amdgcn_wmma_f32_16x16x32_f16(false, pf[c], false, bf,
                                                          (short)0, oacc[dt], false, false);
      }
    __syncthreads();
  }

  // ---- normalize and store context [row, h*64+d] ----
#pragma unroll
  for (int dt = 0; dt < 4; ++dt)
#pragma unroll
    for (int r = 0; r < 8; ++r) {
      int row = qrow0 + r + hi8;
      int d   = dt * 16 + lm;
      float val = oacc[dt][r] / li[r];
      ctx[((bbase + row) * HEADS + h) * (size_t)HDIM + d] = (half_t)val;
    }
}

// ---------- launcher ----------
extern "C" void kernel_launch(void* const* d_in, const int* in_sizes, int n_in,
                              void* d_out, int out_size, void* d_ws, size_t ws_size,
                              hipStream_t stream)
{
  (void)in_sizes; (void)n_in; (void)out_size; (void)ws_size;
  const float* x  = (const float*)d_in[0];
  const float* Wq = (const float*)d_in[1];
  const float* bq = (const float*)d_in[2];
  const float* Wk = (const float*)d_in[3];
  const float* bk = (const float*)d_in[4];
  const float* Wv = (const float*)d_in[5];
  const float* bv = (const float*)d_in[6];
  const float* Wo = (const float*)d_in[7];
  const float* bo = (const float*)d_in[8];
  float* out = (float*)d_out;

  half_t* ws = (half_t*)d_ws;
  size_t off = 0;
  auto carve = [&](size_t n) { half_t* p = ws + off; off += (n + 127) & ~(size_t)127; return p; };
  half_t* xh  = carve((size_t)MTOT * HIDDEN);
  half_t* Wqh = carve((size_t)HIDDEN * HIDDEN);
  half_t* Wkh = carve((size_t)HIDDEN * HDIM);
  half_t* Wvh = carve((size_t)HIDDEN * HDIM);
  half_t* Woh = carve((size_t)HIDDEN * HIDDEN);
  half_t* qh  = carve((size_t)MTOT * HIDDEN);
  half_t* kh  = carve((size_t)MTOT * HDIM);
  half_t* vh  = carve((size_t)MTOT * HDIM);
  half_t* cxh = carve((size_t)MTOT * HIDDEN);

  cvt_f32_f16<<<512, 256, 0, stream>>>(x,  xh,  MTOT * HIDDEN);
  cvt_f32_f16<<<512, 256, 0, stream>>>(Wq, Wqh, HIDDEN * HIDDEN);
  cvt_f32_f16<<<64,  256, 0, stream>>>(Wk, Wkh, HIDDEN * HDIM);
  cvt_f32_f16<<<64,  256, 0, stream>>>(Wv, Wvh, HIDDEN * HDIM);
  cvt_f32_f16<<<512, 256, 0, stream>>>(Wo, Woh, HIDDEN * HIDDEN);

  // projections
  gemm_f16_wmma<<<dim3(MTOT / TM, HIDDEN / TN), 256, 0, stream>>>(
      xh, Wqh, bq, qh, nullptr, MTOT, HIDDEN, HIDDEN, 0);
  gemm_f16_wmma<<<dim3(MTOT / TM, HDIM / TN), 256, 0, stream>>>(
      xh, Wkh, bk, kh, nullptr, MTOT, HDIM, HIDDEN, 0);
  gemm_f16_wmma<<<dim3(MTOT / TM, HDIM / TN), 256, 0, stream>>>(
      xh, Wvh, bv, vh, nullptr, MTOT, HDIM, HIDDEN, 0);

  // flash multi-query attention
  mqa_flash_wmma<<<dim3(SEQ / 64, HEADS, BSZ), 128, 0, stream>>>(qh, kh, vh, cxh, SEQ);

  // output projection -> fp32 d_out
  gemm_f16_wmma<<<dim3(MTOT / TM, HIDDEN / TN), 256, 0, stream>>>(
      cxh, Woh, bo, nullptr, out, MTOT, HIDDEN, HIDDEN, 1);
}
